// RWKV_TimeMix_79207786873576
// MI455X (gfx1250) — matrix operand order repaired and checked by reading, not compiled, not run on hardware
//
#include <hip/hip_runtime.h>
#include <hip/hip_bf16.h>
#include <stdint.h>

// ---------------------------------------------------------------------------
// RWKV TimeMix, MI455X (gfx1250, wave32, WMMA + async global->LDS staging).
// w[h,t,u] = beta[h,t] * wu[h,u] (rank-1)  =>  wkv = beta * S, S = sum wu*k*v.
// Pipeline:
//   1) fp32 -> bf16 convert (x, Wk, Wv, Wr, Wo)
//   2) WMMA bf16 GEMMs (double-buffered async LDS staging): k, v, r
//   3) chunked scan: partial sums -> chunk-offset scan -> fused cumsum+output
//      (cumsum is regenerated on the fly; never materialized in HBM)
//   4) WMMA bf16 GEMM: out = rwkv @ Wo^T + bo
// ---------------------------------------------------------------------------

typedef __bf16 bf16_t;
typedef __attribute__((ext_vector_type(16))) __bf16 v16bf;
typedef __attribute__((ext_vector_type(8)))  __bf16 v8bf;
typedef __attribute__((ext_vector_type(8)))  float  v8f;
typedef __attribute__((ext_vector_type(4)))  int    v4i;

typedef __attribute__((address_space(1))) v4i as1_v4i;   // global
typedef __attribute__((address_space(3))) v4i as3_v4i;   // LDS

constexpr int kB   = 8;
constexpr int kT   = 2048;
constexpr int kC   = 1024;
constexpr int kH   = 16;
constexpr int kHS  = kC / kH;     // 64
constexpr int kM   = kB * kT;     // 16384
constexpr int kCH  = 16;          // scan chunks along T
constexpr int kTCH = kT / kCH;    // 128

// ---- CDNA5 async global->LDS path (ASYNCcnt), with safe fallback ----------
#if __has_builtin(__builtin_amdgcn_global_load_async_to_lds_b128) && \
    __has_builtin(__builtin_amdgcn_s_wait_asynccnt)
#define USE_ASYNC_LDS 1
#else
#define USE_ASYNC_LDS 0
#endif

__device__ __forceinline__ void async_copy_b128(void* lds_dst, const void* gsrc)
{
#if USE_ASYNC_LDS
    as1_v4i* g = (as1_v4i*)(unsigned long long)(uintptr_t)gsrc;
    as3_v4i* l = (as3_v4i*)(unsigned int)(uintptr_t)lds_dst;
    __builtin_amdgcn_global_load_async_to_lds_b128(g, l, 0, 0);
#else
    *(v8bf*)lds_dst = *(const v8bf*)gsrc;
#endif
}

// ------------------------------- convert -----------------------------------
__global__ __launch_bounds__(256) void cvt_f32_to_bf16(const float* __restrict__ src,
                                                       bf16_t* __restrict__ dst,
                                                       int n)
{
    int i = (blockIdx.x * 256 + threadIdx.x) * 4;
    if (i + 3 < n) {
        float4 f = *(const float4*)(src + i);
        dst[i + 0] = (bf16_t)f.x;
        dst[i + 1] = (bf16_t)f.y;
        dst[i + 2] = (bf16_t)f.z;
        dst[i + 3] = (bf16_t)f.w;
    } else {
        for (int j = i; j < n; ++j) dst[j] = (bf16_t)src[j];
    }
}

// ------------------------------- WMMA GEMM ---------------------------------
// Y[M,N] = A[M,K] @ W[N,K]^T + bias[N]
// Workgroup tile 64(M) x 256(N), 8 waves 2x4, wave tile 32x64 (2x4 WMMA tiles).
// K-step 32, double-buffered async LDS staging, last iteration peeled.
#define TM 64
#define TN 256
#define TK 32
#define LDSP (TK + 8)   // padded row stride: 80B, keeps b128 LDS ops aligned

__global__ __launch_bounds__(256) void gemm_bf16_wmma(const bf16_t* __restrict__ A,
                                                      const bf16_t* __restrict__ W,
                                                      const float*  __restrict__ bias,
                                                      float* __restrict__ Y,
                                                      int M, int N, int K)
{
    __shared__ bf16_t Al[2][TM][LDSP];
    __shared__ bf16_t Bl[2][TN][LDSP];

    const int tid  = threadIdx.x;
    const int lane = tid & 31;
    const int wave = tid >> 5;
    const int wm   = wave & 1;   // 0..1 : M sub-block (32 rows)
    const int wn   = wave >> 1;  // 0..3 : N sub-block (64 cols)
    const int m0   = blockIdx.y * TM;
    const int n0   = blockIdx.x * TN;

    v8f acc[2][4] = {};

    const int lrow16  = lane & 15;
    const int a_basek = (lane < 16) ? 0 : 8;    // 16-bit A-matrix lane striping
    const int b_basek = (lane < 16) ? 0 : 8;    // 16-bit B-matrix lane striping
    const int rowoff  = (lane < 16) ? 0 : 8;    // C/D layout

    const int ar = tid >> 2;          // 0..63
    const int ac = (tid & 3) * 8;     // 0,8,16,24

    auto stage = [&](int k0, int buf) {
        async_copy_b128(&Al[buf][ar][ac],
                        A + (size_t)(m0 + ar) * K + k0 + ac);
#pragma unroll
        for (int i = 0; i < 4; ++i) {
            int idx = tid + i * 256;
            int r   = idx >> 2;          // 0..255
            int cc  = (idx & 3) * 8;
            async_copy_b128(&Bl[buf][r][cc],
                            W + (size_t)(n0 + r) * K + k0 + cc);
        }
    };

    auto compute = [&](int buf) {
        v16bf afr[2], bfr[4];
#pragma unroll
        for (int mi = 0; mi < 2; ++mi) {
            const bf16_t* p = &Al[buf][wm * 32 + mi * 16 + lrow16][0];
            v8bf lo = *(const v8bf*)(p + a_basek);
            v8bf hi = *(const v8bf*)(p + a_basek + 16);
            afr[mi] = __builtin_shufflevector(lo, hi, 0,1,2,3,4,5,6,7,8,9,10,11,12,13,14,15);
        }
#pragma unroll
        for (int ni = 0; ni < 4; ++ni) {
            const bf16_t* p = &Bl[buf][wn * 64 + ni * 16 + lrow16][0];
            v8bf lo = *(const v8bf*)(p + b_basek);
            v8bf hi = *(const v8bf*)(p + b_basek + 16);
            bfr[ni] = __builtin_shufflevector(lo, hi, 0,1,2,3,4,5,6,7,8,9,10,11,12,13,14,15);
        }
#pragma unroll
        for (int mi = 0; mi < 2; ++mi)
#pragma unroll
            for (int ni = 0; ni < 4; ++ni)
                acc[mi][ni] = __builtin_amdgcn_wmma_f32_16x16x32_bf16(
                    false, afr[mi], false, bfr[ni], (short)0, acc[mi][ni], false, false);
    };

    const int nk = K / TK;
    stage(0, 0);

    int it = 0;
    for (; it < nk - 1; ++it) {          // steady state: branch-free body
        const int cur = it & 1;
        stage((it + 1) * TK, cur ^ 1);   // overlap next-tile DMA with compute
#if USE_ASYNC_LDS
        __builtin_amdgcn_s_wait_asynccnt(5);   // tile `it` landed; next 5 in flight
#endif
        __syncthreads();
        compute(cur);
        __syncthreads();                 // buffer consumed; safe to overwrite
    }
    // drain last tile
#if USE_ASYNC_LDS
    __builtin_amdgcn_s_wait_asynccnt(0);
#endif
    __syncthreads();
    compute(it & 1);

    // --- epilogue: bias add + fp32 store ---
#pragma unroll
    for (int mi = 0; mi < 2; ++mi) {
#pragma unroll
        for (int ni = 0; ni < 4; ++ni) {
            int col   = n0 + wn * 64 + ni * 16 + lrow16;
            float bv  = bias[col];
            int rbase = m0 + wm * 32 + mi * 16 + rowoff;
#pragma unroll
            for (int r = 0; r < 8; ++r)
                Y[(size_t)(rbase + r) * N + col] = acc[mi][ni][r] + bv;
        }
    }
}

// ------------------------- chunked scan: phase A ---------------------------
// One thread per (b, c, chunk): partial sum of k over the chunk and partial
// S contribution sum(wu[t]*k*v).  131072 threads -> latency fully hidden.
__global__ __launch_bounds__(256) void scan_partials(const float* __restrict__ k,
                                                     const float* __restrict__ v,
                                                     const float* __restrict__ time_w, // H*T
                                                     const float* __restrict__ alpha,  // H*T
                                                     float* __restrict__ Csum,  // (B*CH, C)
                                                     float* __restrict__ Spar)  // (B*CH, C)
{
    int idx   = blockIdx.x * 256 + threadIdx.x;   // 0 .. B*C*CH-1
    int c     = idx % kC;
    int rest  = idx / kC;
    int chunk = rest % kCH;
    int b     = rest / kCH;
    int h     = c / kHS;

    const int t0 = chunk * kTCH;
    float ks = 0.f, sp = 0.f;
    size_t base = ((size_t)b * kT + t0) * kC + c;
    for (int dt = 0; dt < kTCH; ++dt) {
        int t = t0 + dt;
        float kt = k[base + (size_t)dt * kC];
        float vt = v[base + (size_t)dt * kC];
        float wu = time_w[h * kT + t] * alpha[h * kT + t];   // uniform: L2 broadcast
        ks += kt;
        sp += wu * (kt * vt);
    }
    size_t o = ((size_t)b * kCH + chunk) * kC + c;
    Csum[o] = ks;
    Spar[o] = sp;
}

// ------------------------- chunked scan: phase B ---------------------------
// One thread per (b,c): exclusive-scan the 16 chunk sums in place and reduce
// the S partials.  Tiny (8192 threads, 1MB of traffic).
__global__ __launch_bounds__(256) void scan_offsets(float* __restrict__ Csum,  // in/out
                                                    const float* __restrict__ Spar,
                                                    float* __restrict__ S)     // B*C
{
    int gid = blockIdx.x * 256 + threadIdx.x;     // 0 .. B*C-1
    int c = gid % kC;
    int b = gid / kC;

    float run = 0.f, s = 0.f;
#pragma unroll
    for (int ch = 0; ch < kCH; ++ch) {
        size_t o = ((size_t)b * kCH + ch) * kC + c;
        float cs = Csum[o];
        Csum[o] = run;        // exclusive prefix
        run += cs;
        s += Spar[o];
    }
    S[(size_t)b * kC + c] = s;
}

// ------------------- chunked scan: phase C + elementwise -------------------
// One thread per (b, c, chunk): regenerate running cumsum from the chunk
// offset and emit rwkv_bf16 = sigmoid(r) * beta[h,t] * S[b,c] / cumsum.
// cumsum is never written to HBM.
__global__ __launch_bounds__(256) void scan_emit(const float* __restrict__ k,
                                                 const float* __restrict__ r,
                                                 const float* __restrict__ Csum, // offsets
                                                 const float* __restrict__ S,
                                                 const float* __restrict__ beta, // H*T
                                                 bf16_t* __restrict__ out)
{
    int idx   = blockIdx.x * 256 + threadIdx.x;
    int c     = idx % kC;
    int rest  = idx / kC;
    int chunk = rest % kCH;
    int b     = rest / kCH;
    int h     = c / kHS;

    const int t0 = chunk * kTCH;
    float run = Csum[((size_t)b * kCH + chunk) * kC + c];
    float Sv  = S[(size_t)b * kC + c];
    size_t base = ((size_t)b * kT + t0) * kC + c;
    for (int dt = 0; dt < kTCH; ++dt) {
        int t = t0 + dt;
        size_t i = base + (size_t)dt * kC;
        run += k[i];
        float rv  = r[i];
        float sig = 1.f / (1.f + __expf(-rv));
        float val = sig * beta[h * kT + t] * Sv / run;
        out[i] = (bf16_t)val;
    }
}

// ------------------------------- launcher ----------------------------------
extern "C" void kernel_launch(void* const* d_in, const int* in_sizes, int n_in,
                              void* d_out, int out_size, void* d_ws, size_t ws_size,
                              hipStream_t stream)
{
    const float* x       = (const float*)d_in[0];
    const float* time_w  = (const float*)d_in[1];
    const float* alpha   = (const float*)d_in[2];
    const float* beta    = (const float*)d_in[3];
    const float* Wk      = (const float*)d_in[4];
    const float* bk      = (const float*)d_in[5];
    const float* Wv      = (const float*)d_in[6];
    const float* bv      = (const float*)d_in[7];
    const float* Wr      = (const float*)d_in[8];
    const float* br      = (const float*)d_in[9];
    const float* Wo      = (const float*)d_in[10];
    const float* bo      = (const float*)d_in[11];
    float* out           = (float*)d_out;

    size_t off = 0;
    auto carve = [&](size_t bytes) {
        void* p = (char*)d_ws + off;
        off += (bytes + 255) & ~(size_t)255;
        return p;
    };
    const size_t BTC = (size_t)kB * kT * kC;
    const size_t CC  = (size_t)kC * kC;
    const size_t BCH = (size_t)kB * kCH * kC;   // chunk-partial grids

    bf16_t* xbf    = (bf16_t*)carve(BTC * 2);
    bf16_t* Wkbf   = (bf16_t*)carve(CC * 2);
    bf16_t* Wvbf   = (bf16_t*)carve(CC * 2);
    bf16_t* Wrbf   = (bf16_t*)carve(CC * 2);
    bf16_t* Wobf   = (bf16_t*)carve(CC * 2);
    float*  kbuf   = (float*)carve(BTC * 4);
    float*  vbuf   = (float*)carve(BTC * 4);
    float*  rbuf   = (float*)carve(BTC * 4);
    float*  Csum   = (float*)carve(BCH * 4);
    float*  Spar   = (float*)carve(BCH * 4);
    float*  Sbuf   = (float*)carve((size_t)kB * kC * 4);
    bf16_t* rwkvbf = (bf16_t*)carve(BTC * 2);

    {
        int n = (int)BTC;
        cvt_f32_to_bf16<<<(n / 4 + 255) / 256, 256, 0, stream>>>(x, xbf, n);
        int nw = (int)CC;
        int gw = (nw / 4 + 255) / 256;
        cvt_f32_to_bf16<<<gw, 256, 0, stream>>>(Wk, Wkbf, nw);
        cvt_f32_to_bf16<<<gw, 256, 0, stream>>>(Wv, Wvbf, nw);
        cvt_f32_to_bf16<<<gw, 256, 0, stream>>>(Wr, Wrbf, nw);
        cvt_f32_to_bf16<<<gw, 256, 0, stream>>>(Wo, Wobf, nw);
    }

    dim3 ggrid(kC / TN, kM / TM);   // (4, 256)
    gemm_bf16_wmma<<<ggrid, 256, 0, stream>>>(xbf, Wkbf, bk, kbuf, kM, kC, kC);
    gemm_bf16_wmma<<<ggrid, 256, 0, stream>>>(xbf, Wvbf, bv, vbuf, kM, kC, kC);
    gemm_bf16_wmma<<<ggrid, 256, 0, stream>>>(xbf, Wrbf, br, rbuf, kM, kC, kC);

    const unsigned gscan = (unsigned)(((size_t)kB * kC * kCH) / 256);   // 512
    scan_partials<<<gscan, 256, 0, stream>>>(kbuf, vbuf, time_w, alpha, Csum, Spar);
    scan_offsets<<<(kB * kC) / 256, 256, 0, stream>>>(Csum, Spar, Sbuf);
    scan_emit<<<gscan, 256, 0, stream>>>(kbuf, rbuf, Csum, Sbuf, beta, rwkvbf);

    gemm_bf16_wmma<<<ggrid, 256, 0, stream>>>(rwkvbf, Wobf, bo, out, kM, kC, kC);
}
